// PINN_DiffSorp_85478439125427
// MI455X (gfx1250) — matrix operand order repaired and checked by reading, not compiled, not run on hardware
//
#include <hip/hip_runtime.h>
#include <hip/hip_bf16.h>

typedef __attribute__((ext_vector_type(16))) _Float16 v16h;
typedef __attribute__((ext_vector_type(8)))  _Float16 v8h;
typedef __attribute__((ext_vector_type(8)))  float    v8f;

#define LDS_STRIDE 136   // 128 + 8 halfs pad; row stride 272B = 17*16B (keeps b128 alignment)

// ---------------------------------------------------------------------------
// Prep: convert the four 128x128 fp32 hidden-layer weights into f16 fragments
// laid out as the B operand of V_WMMA_F32_16X16X32_F16, in this file's contiguous k order:
//   fragment (kc, ct): lane L holds column n = ct*16 + L%16,
//   element e -> k = kc*32 + (L>=16 ? 16 : 0) + e   (16 halfs contiguous/lane)
// ---------------------------------------------------------------------------
__global__ __launch_bounds__(256) void pinn_prep_weights(
    const float* __restrict__ W1, const float* __restrict__ W2,
    const float* __restrict__ W3, const float* __restrict__ W4,
    _Float16* __restrict__ wf)
{
  int i = blockIdx.x * 256 + threadIdx.x;
  if (i >= 4 * 16384) return;
  int l    = i >> 14;
  int r    = i & 16383;
  int e    = r & 15;
  int lane = (r >> 4) & 31;
  int ct   = (r >> 9) & 7;
  int kc   = (r >> 12) & 3;
  int n = ct * 16 + (lane & 15);
  int k = kc * 32 + ((lane >> 4) << 4) + e;
  const float* W = (l == 0) ? W1 : (l == 1) ? W2 : (l == 2) ? W3 : W4;
  wf[i] = (_Float16)W[k * 128 + n];
}

// ---------------------------------------------------------------------------
// Fused PINN kernel: 8 waves = 4 streams {h, h_t, h_x, h_xx} x 2 row-tiles(16)
// ---------------------------------------------------------------------------
__global__ __launch_bounds__(256) void pinn_fused(
    const float* __restrict__ t,  const float* __restrict__ x,
    const float* __restrict__ W0, const float* __restrict__ b0,
    const _Float16* __restrict__ wf,
    const float* __restrict__ b1, const float* __restrict__ b2,
    const float* __restrict__ b3, const float* __restrict__ b4,
    const float* __restrict__ W5, const float* __restrict__ b5,
    float* __restrict__ out, int N)
{
  __shared__ __align__(16) _Float16 act[4][32][LDS_STRIDE]; // activations per stream
  __shared__ _Float16 sprim[4096];  // primal tanh 'a', fragment-flat layout
  __shared__ _Float16 szx[4096];    // x-stream pre-activation z_x, same layout
  __shared__ float    sfin[32][8];  // u,v, zx_u,zx_v, ut,vt, uxx,vxx

  const int lane   = threadIdx.x & 31;
  const int wave   = threadIdx.x >> 5;
  const int stream = wave & 3;      // 0=h, 1=h_t, 2=h_x, 3=h_xx
  const int rt     = wave >> 2;     // row tile 0..1
  const int mhi    = lane >> 4;     // 0/1: which half-lane group
  const int nlo    = lane & 15;
  const int row0g  = blockIdx.x * 32 + rt * 16;
  const int idxb   = rt * 2048;

  v8f acc[8];                       // 8 column tiles x 8 f32 (C/D layout)

  // Epilogue shared by layers 0..4: tanh + tangent chain rules via LDS coupling
  auto epilogue = [&]() {
    __syncthreads();                                  // all GEMM reads of act done
    if (stream == 0) {
      #pragma unroll
      for (int ct = 0; ct < 8; ++ct)
        #pragma unroll
        for (int v = 0; v < 8; ++v) {
          float a = tanhf(acc[ct][v]);
          sprim[idxb + (ct * 8 + v) * 32 + lane] = (_Float16)a;
          act[0][rt * 16 + v + mhi * 8][ct * 16 + nlo] = (_Float16)a;
        }
    } else if (stream == 2) {
      #pragma unroll
      for (int ct = 0; ct < 8; ++ct)
        #pragma unroll
        for (int v = 0; v < 8; ++v)
          szx[idxb + (ct * 8 + v) * 32 + lane] = (_Float16)acc[ct][v];
    }
    __syncthreads();
    if (stream != 0) {
      #pragma unroll
      for (int ct = 0; ct < 8; ++ct)
        #pragma unroll
        for (int v = 0; v < 8; ++v) {
          int fi = idxb + (ct * 8 + v) * 32 + lane;
          float a   = (float)sprim[fi];
          float fac = 1.0f - a * a;
          float h;
          if (stream == 3) {
            float zx = (float)szx[fi];
            h = fac * acc[ct][v] - 2.0f * a * fac * zx * zx;
          } else {
            h = fac * acc[ct][v];
          }
          act[stream][rt * 16 + v + mhi * 8][ct * 16 + nlo] = (_Float16)h;
        }
    }
    __syncthreads();                                  // act ready for next layer
  };

  // ---------------- Layer 0: input dim 2 -> 128 (plain VALU) ----------------
  {
    float tv[8], xv[8];
    if (stream == 0) {
      #pragma unroll
      for (int v = 0; v < 8; ++v) {
        int r = row0g + v + mhi * 8;
        tv[v] = t[r] * 1e-3f;                         // reference feeds t/1000
        xv[v] = x[r];
      }
    }
    #pragma unroll
    for (int ct = 0; ct < 8; ++ct) {
      int col = ct * 16 + nlo;
      float w0a = W0[col];
      float w0b = W0[128 + col];
      float bb  = b0[col];
      #pragma unroll
      for (int v = 0; v < 8; ++v) {
        float z;
        if      (stream == 0) z = tv[v] * w0a + xv[v] * w0b + bb;
        else if (stream == 1) z = w0a * 1e-3f;        // d(input)/dt = [1e-3, 0]
        else if (stream == 2) z = w0b;                // d(input)/dx = [0, 1]
        else                  z = 0.0f;               // second derivative of input
        acc[ct][v] = z;
      }
    }
    epilogue();
  }

  // ---------------- Layers 1..4: 128->128 via WMMA f16 ----------------------
  #pragma unroll
  for (int l = 0; l < 4; ++l) {
    const _Float16* Wl = wf + l * 16384;
    const float*    bl = (l == 0) ? b1 : (l == 1) ? b2 : (l == 2) ? b3 : b4;

    #pragma unroll
    for (int ct = 0; ct < 8; ++ct) {
      float bb = (stream == 0) ? bl[ct * 16 + nlo] : 0.0f;
      #pragma unroll
      for (int v = 0; v < 8; ++v) acc[ct][v] = bb;
    }

    const _Float16* abase = &act[stream][rt * 16 + nlo][0];
    #pragma unroll
    for (int kc = 0; kc < 4; ++kc) {
      // A fragment (16x32 f16): elems 0..7 -> K = kc*32 + mhi*16 + i,
      //                         elems 8..15 -> K = +8 more
      const _Float16* ap = abase + kc * 32 + mhi * 16;
      v8h lo = *(const v8h*)ap;
      v8h hi = *(const v8h*)(ap + 8);
      v16h A;
      #pragma unroll
      for (int i = 0; i < 8; ++i) { A[i] = lo[i]; A[i + 8] = hi[i]; }

      #pragma unroll
      for (int ct = 0; ct < 8; ++ct) {
        v16h B = *(const v16h*)(Wl + ((kc * 8 + ct) * 32 + lane) * 16);
        acc[ct] = __builtin_amdgcn_wmma_f32_16x16x32_f16(
            false, A, false, B, (short)0, acc[ct], false, false);
      }
    }
    epilogue();
  }

  // ---------------- Layer 5: 128 -> 2, sigmoid, PDE residuals ---------------
  {
    const int row  = nlo;         // 16 rows per tile
    const int csel = mhi;         // 0 -> u column, 1 -> v column
    const int r    = row0g + row;
    const int srow = rt * 16 + row;

    const _Float16* ar = &act[stream][srow][0];
    float z5 = (stream == 0) ? b5[csel] : 0.0f;
    #pragma unroll
    for (int k8 = 0; k8 < 16; ++k8) {
      v8h av = *(const v8h*)(ar + k8 * 8);
      #pragma unroll
      for (int j = 0; j < 8; ++j)
        z5 += (float)av[j] * W5[(k8 * 8 + j) * 2 + csel];
    }

    if (stream == 0) {
      float s = 1.0f / (1.0f + expf(-z5));
      sfin[srow][csel] = s;
      out[csel * N + r] = s;                          // u, v
    } else if (stream == 2) {
      sfin[srow][2 + csel] = z5;                      // z_x at output
    }
    __syncthreads();
    if (stream == 1) {
      float s = sfin[srow][csel];
      sfin[srow][4 + csel] = s * (1.0f - s) * z5;     // u_t, v_t
    } else if (stream == 3) {
      float s  = sfin[srow][csel];
      float zx = sfin[srow][2 + csel];
      float sp = s * (1.0f - s);
      sfin[srow][6 + csel] = sp * (1.0f - 2.0f * s) * zx * zx + sp * z5; // u_xx, v_xx
    }
    __syncthreads();
    if (stream == 3 && csel == 0) {
      float u   = sfin[srow][0];
      float vv  = sfin[srow][1];
      float ut  = sfin[srow][4];
      float vt  = sfin[srow][5];
      float uxx = sfin[srow][6];
      (void)vv;
      const float RET_C = ((1.0f - 0.29f) / 0.29f) * 2880.0f * (1.016f / 2880.0f) * 0.874f;
      float ret = 1.0f + RET_C * powf(u + 1e-6f, 0.874f - 1.0f);
      float f = ut - (0.0005f / ret) * uxx;
      float g = sfin[srow][5] - (0.0005f * 0.29f / (2880.0f / 1000.0f)) * uxx;
      (void)vt;
      out[2 * N + r] = f;
      out[3 * N + r] = g;
    }
  }
}

// ---------------------------------------------------------------------------
extern "C" void kernel_launch(void* const* d_in, const int* in_sizes, int n_in,
                              void* d_out, int out_size, void* d_ws, size_t ws_size,
                              hipStream_t stream) {
  const float* t  = (const float*)d_in[0];
  const float* x  = (const float*)d_in[1];
  const float* W0 = (const float*)d_in[2];
  const float* b0 = (const float*)d_in[3];
  const float* W1 = (const float*)d_in[4];
  const float* b1 = (const float*)d_in[5];
  const float* W2 = (const float*)d_in[6];
  const float* b2 = (const float*)d_in[7];
  const float* W3 = (const float*)d_in[8];
  const float* b3 = (const float*)d_in[9];
  const float* W4 = (const float*)d_in[10];
  const float* b4 = (const float*)d_in[11];
  const float* W5 = (const float*)d_in[12];
  const float* b5 = (const float*)d_in[13];
  _Float16* wf = (_Float16*)d_ws;       // 4 * 16384 halfs = 128 KB of scratch
  int N = in_sizes[0];

  pinn_prep_weights<<<(4 * 16384 + 255) / 256, 256, 0, stream>>>(W1, W2, W3, W4, wf);
  pinn_fused<<<N / 32, 256, 0, stream>>>(t, x, W0, b0, wf,
                                         b1, b2, b3, b4, W5, b5,
                                         (float*)d_out, N);
}
